// EnhancedTrustTemporalGNN_76836964926261
// MI455X (gfx1250) — matrix operand order repaired and checked by reading, not compiled, not run on hardware
//
#include <hip/hip_runtime.h>
#include <cstdint>
#include <cstddef>

// ---------------------------------------------------------------------------
// Types for CDNA5 WMMA (wave32, 16x16x32 f16 -> f32)
// ---------------------------------------------------------------------------
typedef __attribute__((ext_vector_type(16))) _Float16 v16h;
typedef __attribute__((ext_vector_type(8)))  float    v8f;
typedef __attribute__((ext_vector_type(8)))  unsigned uvec8;

__device__ inline unsigned pack2h(float a, float b) {
  union { _Float16 h[2]; unsigned u; } x;
  x.h[0] = (_Float16)a; x.h[1] = (_Float16)b;
  return x.u;
}

__device__ inline v16h cast_frag(uvec8 u) {
  union { uvec8 u; v16h h; } c; c.u = u; return c.h;
}

// ordered-uint encoding of float so unsigned atomicMax == float max
__device__ inline unsigned f2key(float f) {
  unsigned u = __float_as_uint(f);
  return (u & 0x80000000u) ? ~u : (u | 0x80000000u);
}
__device__ inline float key2f(unsigned k) {
  unsigned u = (k & 0x80000000u) ? (k & 0x7FFFFFFFu) : ~k;
  return __uint_as_float(u);
}

// ---------------------------------------------------------------------------
// WMMA GEMM:  C[M x 128] = A[M x Kd] @ W[128 x Kd]^T + bias[128]
// Block = 128 threads (4 waves). Block tile: 64 rows x 128 cols.
// Wave w: rows 16w..16w+15, all 128 cols (8 WMMA n-tiles), K chunks of 32.
// ---------------------------------------------------------------------------
__global__ __launch_bounds__(128) void gemm_bias_wmma(
    const float* __restrict__ A, const float* __restrict__ W,
    const float* __restrict__ bias, float* __restrict__ C,
    int M, int Kd)
{
  // packed pairs of f16: As32[m][k/2], Bs32[n][k/2]  (k within 32-chunk)
  __shared__ unsigned As32[64][17];
  __shared__ unsigned Bs32[128][17];

  const int tid   = threadIdx.x;
  const int wave  = tid >> 5;
  const int lane  = tid & 31;
  const int l15   = lane & 15;
  const int hhalf = lane >> 4;
  const int blockRow = blockIdx.x * 64;

  v8f acc[8];
#pragma unroll
  for (int i = 0; i < 8; ++i) acc[i] = (v8f)0.0f;

  for (int kb = 0; kb < Kd; kb += 32) {
    // stage A tile: 64 rows x 32 k (512 float4 loads / 128 threads)
#pragma unroll
    for (int i = 0; i < 4; ++i) {
      int id = tid + i * 128;
      int r  = id >> 3;
      int c4 = (id & 7) * 4;
      float4 a4 = make_float4(0.f, 0.f, 0.f, 0.f);
      int row = blockRow + r;
      if (row < M) a4 = *(const float4*)(A + (size_t)row * Kd + kb + c4);
      As32[r][(c4 >> 1) + 0] = pack2h(a4.x, a4.y);
      As32[r][(c4 >> 1) + 1] = pack2h(a4.z, a4.w);
    }
    // stage B tile: Bs32[n][k/2] = W[n][kb+k]  (out-dim is always 128)
#pragma unroll
    for (int i = 0; i < 8; ++i) {
      int id = tid + i * 128;
      int n  = id >> 3;
      int c4 = (id & 7) * 4;
      float4 w4 = *(const float4*)(W + (size_t)n * Kd + kb + c4);
      Bs32[n][(c4 >> 1) + 0] = pack2h(w4.x, w4.y);
      Bs32[n][(c4 >> 1) + 1] = pack2h(w4.z, w4.w);
    }
    __syncthreads();

    // A fragment (ISA 7.12.2: 16-bit A 16x32; lanes16-31 shift K by +8)
    uvec8 au;
    const int arow = wave * 16 + l15;
#pragma unroll
    for (int vv = 0; vv < 8; ++vv) {
      int k2 = (vv < 4 ? vv : 4 + vv) + 4 * hhalf;
      au[vv] = As32[arow][k2];
    }
    v16h afrag = cast_frag(au);

#pragma unroll
    for (int nt = 0; nt < 8; ++nt) {
      uvec8 bu;
      const int brow = nt * 16 + l15;
#pragma unroll
      for (int vv = 0; vv < 8; ++vv) bu[vv] = Bs32[brow][(vv < 4 ? vv : 4 + vv) + 4 * hhalf];
      v16h bfrag = cast_frag(bu);
      acc[nt] = __builtin_amdgcn_wmma_f32_16x16x32_f16(
          false, afrag, false, bfrag, (short)0, acc[nt], false, false);
    }
    __syncthreads();
  }

  // store D: lane l, VGPR r -> C[16w + r + 8h][16nt + (l&15)]
#pragma unroll
  for (int nt = 0; nt < 8; ++nt) {
    int col = nt * 16 + l15;
    float bv = bias[col];
#pragma unroll
    for (int r = 0; r < 8; ++r) {
      int row = blockRow + wave * 16 + r + 8 * hhalf;
      if (row < M) C[(size_t)row * 128 + col] = acc[nt][r] + bv;
    }
  }
}

// ---------------------------------------------------------------------------
// Per-layer algebraic fusion (tiny):
//   Wee[128][64] = ew @ edge_w      (linear∘linear on msg)
//   bee[128]     = ew @ edge_b + eb
//   Ate[128]     = tw @ time_w[:,0];  Bte[128] = tw @ time_b + tb
// grid=128 (out dim), block=64
// ---------------------------------------------------------------------------
__global__ void prep_fuse(
    const float* __restrict__ ew, const float* __restrict__ eb,
    const float* __restrict__ edge_w, const float* __restrict__ edge_b,
    const float* __restrict__ tw, const float* __restrict__ tb,
    const float* __restrict__ time_w, const float* __restrict__ time_b,
    float* __restrict__ Wee, float* __restrict__ bee,
    float* __restrict__ Ate, float* __restrict__ Bte)
{
  int i = blockIdx.x;   // out dim 0..127
  int j = threadIdx.x;  // 0..63
  float acc = 0.f;
  for (int k = 0; k < 128; ++k) acc += ew[i * 128 + k] * edge_w[k * 64 + j];
  Wee[i * 64 + j] = acc;
  if (j == 0) {
    float a = eb[i];
    for (int k = 0; k < 128; ++k) a += ew[i * 128 + k] * edge_b[k];
    bee[i] = a;
  } else if (j == 1) {
    float a = 0.f;
    for (int k = 0; k < 128; ++k) a += tw[i * 128 + k] * time_w[k];
    Ate[i] = a;
  } else if (j == 2) {
    float a = tb[i];
    for (int k = 0; k < 128; ++k) a += tw[i * 128 + k] * time_b[k];
    Bte[i] = a;
  }
}

__global__ void zero_f32(float* __restrict__ p, int n) {
  int i = blockIdx.x * blockDim.x + threadIdx.x;
  if (i < n) p[i] = 0.f;
}

// ---------------------------------------------------------------------------
// Edge attention: one wave per edge (8 waves/block, 8 edges per wave).
// Computes attn[e][h] = trust * scale * <Q[dst], K[src]+ee+te> and atomicMax
// into per-(dst,head) segment max (ordered-uint encoding).
// ---------------------------------------------------------------------------
__global__ __launch_bounds__(256) void edge_attn(
    const int* __restrict__ src, const int* __restrict__ dst,
    const float* __restrict__ t, const float* __restrict__ msg,
    const float* __restrict__ Q, const float* __restrict__ K,
    const float* __restrict__ mem,
    const float* __restrict__ Wee, const float* __restrict__ bee,
    const float* __restrict__ Ate, const float* __restrict__ Bte,
    const float* __restrict__ m1w, const float* __restrict__ m1b,
    const float* __restrict__ m2w, const float* __restrict__ m2b,
    float* __restrict__ attnbuf, unsigned* __restrict__ segmaxU, int E)
{
  __shared__ float sWee[128][65];     // padded: ~2-way conflicts max
  __shared__ float sBee[128], sAte[128], sBte[128];
  __shared__ float sM1w[48], sM1b[16], sM2w[16], sM2b[1];

  const int tid = threadIdx.x;
  for (int i = tid; i < 128 * 64; i += 256) sWee[i >> 6][i & 63] = Wee[i];
  for (int i = tid; i < 128; i += 256) {
    sBee[i] = bee[i]; sAte[i] = Ate[i]; sBte[i] = Bte[i];
  }
  if (tid < 48) sM1w[tid] = m1w[tid];
  if (tid < 16) { sM1b[tid] = m1b[tid]; sM2w[tid] = m2w[tid]; }
  if (tid == 0) sM2b[0] = m2b[0];
  __syncthreads();

  const int lane = tid & 31;
  const int wave = tid >> 5;
  const int d0   = lane * 4;       // this lane owns dims d0..d0+3
  const int head = lane >> 3;      // 32 dims per head -> 8 lanes per head

  for (int it = 0; it < 8; ++it) {
    int e = blockIdx.x * 64 + it * 8 + wave;
    if (e >= E) continue;
    int   sv = src[e];
    int   dv = dst[e];
    float tv = t[e];

    // msg row held 2 floats per lane; broadcast via shfl in the matvec
    float mA = msg[(size_t)e * 64 + lane];
    float mB = msg[(size_t)e * 64 + 32 + lane];

    float e0 = sBee[d0 + 0], e1 = sBee[d0 + 1];
    float e2 = sBee[d0 + 2], e3 = sBee[d0 + 3];
#pragma unroll 8
    for (int k = 0; k < 32; ++k) {
      float mk = __shfl(mA, k, 32);
      e0 += mk * sWee[d0 + 0][k]; e1 += mk * sWee[d0 + 1][k];
      e2 += mk * sWee[d0 + 2][k]; e3 += mk * sWee[d0 + 3][k];
    }
#pragma unroll 8
    for (int k = 0; k < 32; ++k) {
      float mk = __shfl(mB, k, 32);
      e0 += mk * sWee[d0 + 0][32 + k]; e1 += mk * sWee[d0 + 1][32 + k];
      e2 += mk * sWee[d0 + 2][32 + k]; e3 += mk * sWee[d0 + 3][32 + k];
    }

    float4 qi = *(const float4*)(Q   + (size_t)dv * 128 + d0);
    float4 kk = *(const float4*)(K   + (size_t)sv * 128 + d0);
    float4 md = *(const float4*)(mem + (size_t)dv * 128 + d0);
    float4 ms = *(const float4*)(mem + (size_t)sv * 128 + d0);

    float kj0 = kk.x + e0 + tv * sAte[d0 + 0] + sBte[d0 + 0];
    float kj1 = kk.y + e1 + tv * sAte[d0 + 1] + sBte[d0 + 1];
    float kj2 = kk.z + e2 + tv * sAte[d0 + 2] + sBte[d0 + 2];
    float kj3 = kk.w + e3 + tv * sAte[d0 + 3] + sBte[d0 + 3];

    float dotp = qi.x * kj0 + qi.y * kj1 + qi.z * kj2 + qi.w * kj3;
    float qn = qi.x * qi.x + qi.y * qi.y + qi.z * qi.z + qi.w * qi.w;
    float kn = kj0 * kj0 + kj1 * kj1 + kj2 * kj2 + kj3 * kj3;
    float a0 = qi.x - md.x, a1 = qi.y - md.y, a2 = qi.z - md.z, a3 = qi.w - md.w;
    float dr = a0 * a0 + a1 * a1 + a2 * a2 + a3 * a3;
    float b0 = kj0 - ms.x, b1 = kj1 - ms.y, b2 = kj2 - ms.z, b3 = kj3 - ms.w;
    float dvv = b0 * b0 + b1 * b1 + b2 * b2 + b3 * b3;

    // reduce over the 8 lanes of this head
#pragma unroll
    for (int off = 1; off < 8; off <<= 1) {
      dotp += __shfl_xor(dotp, off, 32);
      qn   += __shfl_xor(qn,   off, 32);
      kn   += __shfl_xor(kn,   off, 32);
      dr   += __shfl_xor(dr,   off, 32);
      dvv  += __shfl_xor(dvv,  off, 32);
    }

    float attn  = dotp * 0.17677669529663688f;  // 1/sqrt(32)
    float qnorm = fmaxf(sqrtf(qn), 1e-8f);
    float knorm = fmaxf(sqrtf(kn), 1e-8f);
    float cosv  = dotp / (qnorm * knorm);
    float drift = sqrtf(dr);
    float devi  = sqrtf(dvv);

    // trust MLP 3 -> 16 (exact gelu) -> 1 -> sigmoid
    float z = sM2b[0];
#pragma unroll
    for (int i = 0; i < 16; ++i) {
      float hv = sM1w[i * 3 + 0] * cosv + sM1w[i * 3 + 1] * drift +
                 sM1w[i * 3 + 2] * devi + sM1b[i];
      hv = 0.5f * hv * (1.0f + erff(hv * 0.7071067811865476f));
      z += sM2w[i] * hv;
    }
    float trust = 1.0f / (1.0f + expf(-z));
    attn *= trust;

    if ((lane & 7) == 0) {
      attnbuf[(size_t)e * 4 + head] = attn;
      atomicMax(&segmaxU[(size_t)dv * 4 + head], f2key(attn));
    }
  }
}

// ---------------------------------------------------------------------------
// exp(attn - segmax[dst]) and accumulate segment sums
// ---------------------------------------------------------------------------
__global__ void edge_softmax(
    const int* __restrict__ dst, float* __restrict__ attnbuf,
    const unsigned* __restrict__ segmaxU, float* __restrict__ segsum, int E)
{
  int i = blockIdx.x * blockDim.x + threadIdx.x;
  if (i >= E * 4) return;
  int e  = i >> 2;
  int hh = i & 3;
  int dv = dst[e];
  float m  = key2f(segmaxU[(size_t)dv * 4 + hh]);
  float ex = expf(attnbuf[i] - m);
  attnbuf[i] = ex;
  atomicAdd(&segsum[(size_t)dv * 4 + hh], ex);
}

// ---------------------------------------------------------------------------
// out[dst] += a * V[src]   (wave per edge; float4 gather + 4 atomic adds)
// ---------------------------------------------------------------------------
__global__ __launch_bounds__(256) void edge_aggr(
    const int* __restrict__ src, const int* __restrict__ dst,
    const float* __restrict__ attnbuf, const float* __restrict__ segsum,
    const float* __restrict__ V, float* __restrict__ outagg, int E)
{
  const int lane = threadIdx.x & 31;
  const int wave = threadIdx.x >> 5;
  int e = blockIdx.x * 8 + wave;
  if (e >= E) return;
  int sv = src[e];
  int dv = dst[e];
  int head = lane >> 3;
  float ex = attnbuf[(size_t)e * 4 + head];
  float ss = segsum[(size_t)dv * 4 + head];
  float a  = ex / (ss + 1e-16f);
  int d0 = lane * 4;
  float4 v4 = *(const float4*)(V + (size_t)sv * 128 + d0);
  float* o = outagg + (size_t)dv * 128 + d0;
  atomicAdd(o + 0, v4.x * a);
  atomicAdd(o + 1, v4.y * a);
  atomicAdd(o + 2, v4.z * a);
  atomicAdd(o + 3, v4.w * a);
}

// ---------------------------------------------------------------------------
// Gate + LayerNorm (wave per node)
// ---------------------------------------------------------------------------
__global__ __launch_bounds__(256) void node_update(
    const float* __restrict__ h, const float* __restrict__ outagg,
    const float* __restrict__ gw, const float* __restrict__ gb,
    const float* __restrict__ ln_g, const float* __restrict__ ln_b,
    float* __restrict__ dest, int Nn)
{
  const int lane = threadIdx.x & 31;
  const int wave = threadIdx.x >> 5;
  int n = blockIdx.x * 8 + wave;
  if (n >= Nn) return;
  int d0 = lane * 4;
  float4 hv = *(const float4*)(h      + (size_t)n * 128 + d0);
  float4 ov = *(const float4*)(outagg + (size_t)n * 128 + d0);
  float4 gv = *(const float4*)(gw + d0);
  float gdot = hv.x * gv.x + hv.y * gv.y + hv.z * gv.z + hv.w * gv.w;
#pragma unroll
  for (int off = 1; off < 32; off <<= 1) gdot += __shfl_xor(gdot, off, 32);
  float beta = 1.0f / (1.0f + expf(-(gdot + gb[0])));
  float y0 = beta * hv.x + (1.0f - beta) * ov.x;
  float y1 = beta * hv.y + (1.0f - beta) * ov.y;
  float y2 = beta * hv.z + (1.0f - beta) * ov.z;
  float y3 = beta * hv.w + (1.0f - beta) * ov.w;
  float s = y0 + y1 + y2 + y3;
#pragma unroll
  for (int off = 1; off < 32; off <<= 1) s += __shfl_xor(s, off, 32);
  float mu = s * (1.0f / 128.0f);
  float c0 = y0 - mu, c1 = y1 - mu, c2 = y2 - mu, c3 = y3 - mu;
  float ssq = c0 * c0 + c1 * c1 + c2 * c2 + c3 * c3;
#pragma unroll
  for (int off = 1; off < 32; off <<= 1) ssq += __shfl_xor(ssq, off, 32);
  float rstd = rsqrtf(ssq * (1.0f / 128.0f) + 1e-5f);
  float4 r;
  r.x = c0 * rstd * ln_g[d0 + 0] + ln_b[d0 + 0];
  r.y = c1 * rstd * ln_g[d0 + 1] + ln_b[d0 + 1];
  r.z = c2 * rstd * ln_g[d0 + 2] + ln_b[d0 + 2];
  r.w = c3 * rstd * ln_g[d0 + 3] + ln_b[d0 + 3];
  *(float4*)(dest + (size_t)n * 128 + d0) = r;
}

// ---------------------------------------------------------------------------
// Host launcher.
// Input order (jax pytree leaves, dicts flatten with sorted keys):
//  0:x 1:msg 2:t 3:edge_index 4:memory
//  5:edge_b 6:edge_w 7:in_b 8:in_w
//  layers li in 0..1, base L = 9 + 18*li:
//   +0 eb +1 ew +2 gb +3 gw +4 kb +5 kw +6 ln_b +7 ln_g
//   +8 m1b +9 m1w +10 m2b +11 m2w +12 qb +13 qw +14 tb +15 tw +16 vb +17 vw
//  45:time_b 46:time_w
// ---------------------------------------------------------------------------
extern "C" void kernel_launch(void* const* d_in, const int* in_sizes, int n_in,
                              void* d_out, int out_size, void* d_ws, size_t ws_size,
                              hipStream_t stream)
{
  (void)n_in; (void)out_size; (void)ws_size;
  const int N = in_sizes[0] / 128;
  const int E = in_sizes[2];

  const float* x    = (const float*)d_in[0];
  const float* msg  = (const float*)d_in[1];
  const float* t    = (const float*)d_in[2];
  const int*   ei   = (const int*)  d_in[3];
  const float* mem  = (const float*)d_in[4];
  const int* src = ei;
  const int* dst = ei + E;

  const float* edge_b = (const float*)d_in[5];
  const float* edge_w = (const float*)d_in[6];
  const float* in_b   = (const float*)d_in[7];
  const float* in_w   = (const float*)d_in[8];
  const float* time_b = (const float*)d_in[45];
  const float* time_w = (const float*)d_in[46];

  // workspace layout (floats)
  float* ws = (float*)d_ws;
  size_t off = 0;
  float* h      = ws + off; off += (size_t)N * 128;
  float* Qb     = ws + off; off += (size_t)N * 128;
  float* Kb     = ws + off; off += (size_t)N * 128;
  float* Vb     = ws + off; off += (size_t)N * 128;
  float* outagg = ws + off; off += (size_t)N * 128;   // contiguous with segmax/segsum
  unsigned* segmaxU = (unsigned*)(ws + off); off += (size_t)N * 4;
  float* segsum = ws + off; off += (size_t)N * 4;
  float* attnb  = ws + off; off += (size_t)E * 4;
  float* Wee    = ws + off; off += 128 * 64;
  float* bee    = ws + off; off += 128;
  float* Ate    = ws + off; off += 128;
  float* Bte    = ws + off; off += 128;

  const int gemmGrid = (N + 63) / 64;

  // input embedding: h = x @ in_w^T + in_b
  gemm_bias_wmma<<<gemmGrid, 128, 0, stream>>>(x, in_w, in_b, h, N, 128);

  for (int li = 0; li < 2; ++li) {
    const int L = 9 + li * 18;
    const float* eb   = (const float*)d_in[L + 0];
    const float* ew   = (const float*)d_in[L + 1];
    const float* gb   = (const float*)d_in[L + 2];
    const float* gw   = (const float*)d_in[L + 3];
    const float* kbia = (const float*)d_in[L + 4];
    const float* kw   = (const float*)d_in[L + 5];
    const float* ln_b = (const float*)d_in[L + 6];
    const float* ln_g = (const float*)d_in[L + 7];
    const float* m1b  = (const float*)d_in[L + 8];
    const float* m1w  = (const float*)d_in[L + 9];
    const float* m2b  = (const float*)d_in[L + 10];
    const float* m2w  = (const float*)d_in[L + 11];
    const float* qbia = (const float*)d_in[L + 12];
    const float* qw   = (const float*)d_in[L + 13];
    const float* tb   = (const float*)d_in[L + 14];
    const float* tw   = (const float*)d_in[L + 15];
    const float* vbia = (const float*)d_in[L + 16];
    const float* vw   = (const float*)d_in[L + 17];

    prep_fuse<<<128, 64, 0, stream>>>(ew, eb, edge_w, edge_b,
                                      tw, tb, time_w, time_b,
                                      Wee, bee, Ate, Bte);

    gemm_bias_wmma<<<gemmGrid, 128, 0, stream>>>(h, qw, qbia, Qb, N, 128);
    gemm_bias_wmma<<<gemmGrid, 128, 0, stream>>>(h, kw, kbia, Kb, N, 128);
    gemm_bias_wmma<<<gemmGrid, 128, 0, stream>>>(h, vw, vbia, Vb, N, 128);

    // zero outagg + segmax + segsum (contiguous: N*136 floats)
    int zn = N * 136;
    zero_f32<<<(zn + 255) / 256, 256, 0, stream>>>(outagg, zn);

    edge_attn<<<(E + 63) / 64, 256, 0, stream>>>(
        src, dst, t, msg, Qb, Kb, mem, Wee, bee, Ate, Bte,
        m1w, m1b, m2w, m2b, attnb, segmaxU, E);

    edge_softmax<<<(E * 4 + 255) / 256, 256, 0, stream>>>(
        dst, attnb, segmaxU, segsum, E);

    edge_aggr<<<(E + 7) / 8, 256, 0, stream>>>(
        src, dst, attnb, segsum, Vb, outagg, E);

    float* destp = (li == 1) ? (float*)d_out : h;
    node_update<<<(N + 7) / 8, 256, 0, stream>>>(
        h, outagg, gw, gb, ln_g, ln_b, destp, N);
  }
}
